// SeqAEGMM_27711128994051
// MI455X (gfx1250) — matrix operand order repaired and checked by reading, not compiled, not run on hardware
//
#include <hip/hip_runtime.h>
#include <math.h>

// ---------------------------------------------------------------- constants
#define BB 256
#define SS 512
#define FF 128
#define HH 128
#define KK 8
#define NG 4
#define LDIM 129   // H + 1

typedef __attribute__((ext_vector_type(16))) __bf16 v16bf;
typedef __attribute__((ext_vector_type(8)))  float  v8f;

// float -> bf16 round-to-nearest-even via bits
__device__ inline __bf16 f2bf(float f) {
  unsigned int u = __builtin_bit_cast(unsigned int, f);
  unsigned int r = (u + 0x7FFFu + ((u >> 16) & 1u)) >> 16;
  unsigned short s = (unsigned short)r;
  return __builtin_bit_cast(__bf16, s);
}

__device__ inline float sigmoidf_(float x) { return 1.0f / (1.0f + __expf(-x)); }

// ------------------------------------------------ WMMA fragment helpers
// Swizzled A storage: element (m, k) of a 16x128 activation matrix stored so
// each lane's 16-element A-fragment (per 32-wide k-chunk) is contiguous.
// ISA A layout (16-bit, 16x32): lane<16 holds k = {0..7,16..23}, lane>=16
// holds k = {8..15,24..31} for row m = lane&15.
__device__ inline int sw_idx(int m, int k) {
  int kc = k >> 5;
  int r = k & 31;
  int half = (r >> 3) & 1;
  int e = (r & 7) | (((r >> 4) & 1) << 3);
  int lane = half * 16 + m;
  return (kc * 32 + lane) * 16 + e;
}

struct afrags { v16bf a0, a1, a2, a3; };

// 4 k-chunk A-fragments, each 32 contiguous bytes per lane -> 2x ds_load_b128
__device__ inline afrags load_afrags(const __bf16* __restrict__ Asw, int lane) {
  afrags A;
#pragma unroll
  for (int e = 0; e < 16; ++e) A.a0[e] = Asw[(0 * 32 + lane) * 16 + e];
#pragma unroll
  for (int e = 0; e < 16; ++e) A.a1[e] = Asw[(1 * 32 + lane) * 16 + e];
#pragma unroll
  for (int e = 0; e < 16; ++e) A.a2[e] = Asw[(2 * 32 + lane) * 16 + e];
#pragma unroll
  for (int e = 0; e < 16; ++e) A.a3[e] = Asw[(3 * 32 + lane) * 16 + e];
  return A;
}

// B fragment: 32x16 bf16 from weight matrix W stored [N][K=128] row-major.
__device__ inline v16bf load_bfrag(const __bf16* __restrict__ W, int lane, int nb, int kb) {
  const __bf16* row = W + (size_t)(nb + (lane & 15)) * 128 + kb + ((lane >> 4) ? 8 : 0);
  v16bf v;
#pragma unroll
  for (int e = 0; e < 16; ++e) v[e] = row[e < 8 ? e : e + 8];   // k = {0..7,16..23} (+8 upper half) -> 2x ds_load_b128
  return v;
}

#define WMMA_BF16(a, b, acc) \
  __builtin_amdgcn_wmma_f32_16x16x32_bf16(false, (a), false, (b), (short)0, (acc), false, false)

// one 16x16 n-tile of A(16x128) * W^T, A preloaded
__device__ inline v8f gemm_tile(const afrags& A, const __bf16* __restrict__ W,
                                int lane, int nb) {
  v8f acc = {};
  acc = WMMA_BF16(A.a0, load_bfrag(W, lane, nb, 0), acc);
  acc = WMMA_BF16(A.a1, load_bfrag(W, lane, nb, 32), acc);
  acc = WMMA_BF16(A.a2, load_bfrag(W, lane, nb, 64), acc);
  acc = WMMA_BF16(A.a3, load_bfrag(W, lane, nb, 96), acc);
  return acc;
}

// ---------------------------------------------------------------- reductions
template <int NT>
__device__ inline float block_reduce_sum(float v, float* scratch) {
  int tid = threadIdx.x;
  scratch[tid] = v;
  __syncthreads();
#pragma unroll
  for (int s = NT / 2; s > 0; s >>= 1) {
    if (tid < s) scratch[tid] += scratch[tid + s];
    __syncthreads();
  }
  float r = scratch[0];
  __syncthreads();
  return r;
}

// ============================================================ encoder GRU
// Persistent kernel: 16 blocks x 16 batch rows, weights resident in LDS bf16.
// Wave w owns columns 16w..16w+15 of every gate -> all gate math in registers,
// hidden state in registers (C-fragment layout), ONE barrier per step.
__global__ __launch_bounds__(256) void enc_kernel(
    const float* __restrict__ x, const float* __restrict__ Wih,
    const float* __restrict__ Whh, const float* __restrict__ bih,
    const float* __restrict__ bhh, float* __restrict__ henc) {
  extern __shared__ char smem[];
  __bf16* sWih = (__bf16*)smem;          // [384][128]
  __bf16* sWhh = sWih + 384 * 128;       // [384][128]
  __bf16* sh0  = sWhh + 384 * 128;       // [16][128] swizzled, double buffered
  __bf16* sh1  = sh0 + 16 * 128;
  __bf16* sx0  = sh1 + 16 * 128;         // [16][128] swizzled, double buffered
  __bf16* sx1  = sx0 + 16 * 128;

  int tid = threadIdx.x, lane = tid & 31, wave = tid >> 5;
  int r0 = blockIdx.x * 16;
  int c = wave * 16 + (lane & 15);       // owned gate column
  int mb = (lane >> 4) ? 8 : 0;          // C-fragment row base

  // swizzled store base for column c, rows mb..mb+7 (stride 16 elements)
  int swbase;
  {
    int kc = c >> 5, r = c & 31;
    int half = (r >> 3) & 1;
    int e = (r & 7) | (((r >> 4) & 1) << 3);
    swbase = (kc * 32 + half * 16 + mb) * 16 + e;
  }

  // per-lane gate biases (column-indexed)
  float bir = bih[c],       bhr = bhh[c];
  float biz = bih[128 + c], bhz = bhh[128 + c];
  float bin = bih[256 + c], bhn = bhh[256 + c];

  for (int i = tid; i < 384 * 128; i += 256) {
    sWih[i] = f2bf(Wih[i]);
    sWhh[i] = f2bf(Whh[i]);
  }
  for (int i = tid; i < 16 * 128; i += 256) sh0[i] = f2bf(0.0f);
  // prologue: x[0] into sx0
#pragma unroll
  for (int u = 0; u < 8; ++u) {
    int e = tid * 8 + u;
    int m = e >> 7, cc = e & 127;
    sx0[sw_idx(m, cc)] = f2bf(x[((size_t)(r0 + m) * SS + 0) * FF + cc]);
  }
  v8f hreg = {};
  __syncthreads();

  for (int t = 0; t < SS; ++t) {
    const __bf16* shp = (t & 1) ? sh1 : sh0;
    const __bf16* sxp = (t & 1) ? sx1 : sx0;
    __bf16* shn = (t & 1) ? sh0 : sh1;
    __bf16* sxn = (t & 1) ? sx0 : sx1;

    afrags Ax = load_afrags(sxp, lane);
    afrags Ah = load_afrags(shp, lane);
    v8f ir = gemm_tile(Ax, sWih, lane, c & ~15);
    v8f hr = gemm_tile(Ah, sWhh, lane, c & ~15);
    v8f iz = gemm_tile(Ax, sWih, lane, 128 + (c & ~15));
    v8f hz = gemm_tile(Ah, sWhh, lane, 128 + (c & ~15));
    v8f in_ = gemm_tile(Ax, sWih, lane, 256 + (c & ~15));
    v8f hn = gemm_tile(Ah, sWhh, lane, 256 + (c & ~15));

#pragma unroll
    for (int j = 0; j < 8; ++j) {
      float r = sigmoidf_(ir[j] + bir + hr[j] + bhr);
      float zz = sigmoidf_(iz[j] + biz + hz[j] + bhz);
      float nn = tanhf(in_[j] + bin + r * (hn[j] + bhn));
      float hv = (1.0f - zz) * nn + zz * hreg[j];
      hreg[j] = hv;
      shn[swbase + j * 16] = f2bf(hv);
    }

    if (t + 1 < SS) {  // stream next x tile (block-uniform branch)
#pragma unroll
      for (int u = 0; u < 8; ++u) {
        int e = tid * 8 + u;
        int m = e >> 7, cc = e & 127;
        sxn[sw_idx(m, cc)] = f2bf(x[((size_t)(r0 + m) * SS + (t + 1)) * FF + cc]);
      }
    }
    __syncthreads();
  }

  // final hidden from registers (C-fragment layout: row mb+j, col c)
#pragma unroll
  for (int j = 0; j < 8; ++j)
    henc[(size_t)(r0 + mb + j) * HH + c] = hreg[j];
}

// ============================================================ decoder GRU
__global__ __launch_bounds__(256) void dec_kernel(
    const float* __restrict__ henc, const float* __restrict__ Wih,
    const float* __restrict__ Whh, const float* __restrict__ bih,
    const float* __restrict__ bhh, const float* __restrict__ Wo,
    const float* __restrict__ bo, float* __restrict__ outbuf) {
  extern __shared__ char smem[];
  __bf16* sWih = (__bf16*)smem;          // [384][128]
  __bf16* sWhh = sWih + 384 * 128;       // [384][128]
  __bf16* sWo  = sWhh + 384 * 128;       // [128][128]
  __bf16* sh0  = sWo + 128 * 128;        // [16][128] swizzled, double buffered
  __bf16* sh1  = sh0 + 16 * 128;
  __bf16* so   = sh1 + 16 * 128;         // [16][128] swizzled

  int tid = threadIdx.x, lane = tid & 31, wave = tid >> 5;
  int r0 = blockIdx.x * 16;
  int c = wave * 16 + (lane & 15);
  int mb = (lane >> 4) ? 8 : 0;

  int swbase;
  {
    int kc = c >> 5, r = c & 31;
    int half = (r >> 3) & 1;
    int e = (r & 7) | (((r >> 4) & 1) << 3);
    swbase = (kc * 32 + half * 16 + mb) * 16 + e;
  }

  float bir = bih[c],       bhr = bhh[c];
  float biz = bih[128 + c], bhz = bhh[128 + c];
  float bin = bih[256 + c], bhn = bhh[256 + c];
  float boc = bo[c];

  for (int i = tid; i < 384 * 128; i += 256) {
    sWih[i] = f2bf(Wih[i]);
    sWhh[i] = f2bf(Whh[i]);
  }
  for (int i = tid; i < 128 * 128; i += 256) sWo[i] = f2bf(Wo[i]);

  // initial hidden: registers (C-fragment layout) + swizzled bf16 in sh0
  v8f hreg;
#pragma unroll
  for (int j = 0; j < 8; ++j) {
    float hv = henc[(size_t)(r0 + mb + j) * HH + c];
    hreg[j] = hv;
    sh0[swbase + j * 16] = f2bf(hv);
  }
  __syncthreads();

  for (int i = 0; i < SS; ++i) {
    const __bf16* shp = (i & 1) ? sh1 : sh0;
    __bf16* shn = (i & 1) ? sh0 : sh1;

    afrags Ah = load_afrags(shp, lane);

    // o = h out_W^T + out_b (wave w -> o columns 16w..16w+15)
    v8f o = gemm_tile(Ah, sWo, lane, c & ~15);
    int ti = SS - 1 - i;   // reference reverses decoder time
#pragma unroll
    for (int j = 0; j < 8; ++j) {
      float v = o[j] + boc;
      so[swbase + j * 16] = f2bf(v);
      outbuf[((size_t)(r0 + mb + j) * SS + ti) * FF + c] = v;
    }
    __syncthreads();

    afrags Ao = load_afrags(so, lane);
    v8f ir = gemm_tile(Ao, sWih, lane, c & ~15);
    v8f hr = gemm_tile(Ah, sWhh, lane, c & ~15);
    v8f iz = gemm_tile(Ao, sWih, lane, 128 + (c & ~15));
    v8f hz = gemm_tile(Ah, sWhh, lane, 128 + (c & ~15));
    v8f in_ = gemm_tile(Ao, sWih, lane, 256 + (c & ~15));
    v8f hn = gemm_tile(Ah, sWhh, lane, 256 + (c & ~15));

#pragma unroll
    for (int j = 0; j < 8; ++j) {
      float r = sigmoidf_(ir[j] + bir + hr[j] + bhr);
      float zz = sigmoidf_(iz[j] + biz + hz[j] + bhz);
      float nn = tanhf(in_[j] + bin + r * (hn[j] + bhn));
      float hv = (1.0f - zz) * nn + zz * hreg[j];
      hreg[j] = hv;
      shn[swbase + j * 16] = f2bf(hv);
    }
    __syncthreads();
  }
}

// ============================================================ cosine per row
__global__ __launch_bounds__(256) void cos_kernel(const float* __restrict__ a,
                                                  const float* __restrict__ o,
                                                  float* __restrict__ rc) {
  __shared__ float red[256];
  int b = blockIdx.x, tid = threadIdx.x;
  const float* ar = a + (size_t)b * (SS * FF);
  const float* br = o + (size_t)b * (SS * FF);
  float sab = 0.f, saa = 0.f, sbb = 0.f;
  for (int i = tid; i < SS * FF; i += 256) {
    float xv = ar[i], yv = br[i];
    sab += xv * yv;
    saa += xv * xv;
    sbb += yv * yv;
  }
  sab = block_reduce_sum<256>(sab, red);
  saa = block_reduce_sum<256>(saa, red);
  sbb = block_reduce_sum<256>(sbb, red);
  if (tid == 0)
    rc[b] = sab / (fmaxf(sqrtf(saa), 1e-8f) * fmaxf(sqrtf(sbb), 1e-8f));
}

// ============================================================ prototype LN
__global__ __launch_bounds__(128) void protln_kernel(const float* __restrict__ prot,
                                                     const float* __restrict__ w,
                                                     const float* __restrict__ bset,
                                                     float* __restrict__ out) {
  __shared__ float red[128];
  int k = blockIdx.x, j = threadIdx.x;
  float v = prot[k * HH + j];
  float mu = block_reduce_sum<128>(v, red) * (1.0f / HH);
  float d = v - mu;
  float var = block_reduce_sum<128>(d * d, red) * (1.0f / HH);
  out[k * HH + j] = w[j] * d * rsqrtf(var + 1e-12f) + bset[j];
}

// ================================== clustering / aggregation / estimator
__global__ __launch_bounds__(128) void cluster_kernel(
    const float* __restrict__ henc, const float* __restrict__ protln,
    const float* __restrict__ lnz_w, const float* __restrict__ lnz_b,
    const float* __restrict__ lna_w, const float* __restrict__ lna_b,
    const float* __restrict__ beta, const float* __restrict__ e1W,
    const float* __restrict__ e1b, const float* __restrict__ e2W,
    const float* __restrict__ e2b, const float* __restrict__ rc,
    float* __restrict__ z_out, float* __restrict__ gamma_out,
    float* __restrict__ dis_out) {
  __shared__ float red[128];
  __shared__ float attn[KK];
  __shared__ float h1s[10];
  int b = blockIdx.x, j = threadIdx.x;

  float hj = henc[(size_t)b * HH + j];
  float mu = block_reduce_sum<128>(hj, red) * (1.0f / HH);
  float d = hj - mu;
  float var = block_reduce_sum<128>(d * d, red) * (1.0f / HH);
  float zln = lnz_w[j] * d * rsqrtf(var + 1e-12f) + lnz_b[j];

  // attention logits over K prototypes
  for (int k = 0; k < KK; ++k) {
    float t = block_reduce_sum<128>(zln * protln[k * HH + j], red);
    if (j == 0) attn[k] = t * 0.08838834764831845f;  // 1/sqrt(128)
  }
  __syncthreads();
  if (j == 0) {
    float mx = attn[0];
    for (int k = 1; k < KK; ++k) mx = fmaxf(mx, attn[k]);
    float s = 0.f;
    for (int k = 0; k < KK; ++k) { attn[k] = __expf(attn[k] - mx); s += attn[k]; }
    float inv = 1.0f / s;
    for (int k = 0; k < KK; ++k) attn[k] *= inv;
  }
  __syncthreads();

  // dis[b][k][:] = LN(beta[k] + attn[k] * h), accumulate mean over k
  float acc = 0.f;
  for (int k = 0; k < KK; ++k) {
    float v = beta[k * HH + j] + attn[k] * hj;
    float mk = block_reduce_sum<128>(v, red) * (1.0f / HH);
    float dv = v - mk;
    float vk = block_reduce_sum<128>(dv * dv, red) * (1.0f / HH);
    float disv = lna_w[j] * dv * rsqrtf(vk + 1e-12f) + lna_b[j];
    dis_out[((size_t)b * KK + k) * HH + j] = disv;
    acc += disv;
  }
  float zj = acc * (1.0f / KK);
  z_out[(size_t)b * LDIM + j] = zj;
  float rcb = rc[b];
  if (j == 0) z_out[(size_t)b * LDIM + HH] = rcb;

  // estimator MLP: h1 = tanh(z e1W^T + e1b)
  for (int o = 0; o < 10; ++o) {
    float t = block_reduce_sum<128>(zj * e1W[o * LDIM + j], red);
    if (j == 0) h1s[o] = tanhf(t + rcb * e1W[o * LDIM + HH] + e1b[o]);
  }
  __syncthreads();
  if (j == 0) {
    float lg[NG];
    float mx = -1e30f;
    for (int g = 0; g < NG; ++g) {
      float t = e2b[g];
      for (int o = 0; o < 10; ++o) t += h1s[o] * e2W[g * 10 + o];
      lg[g] = t;
      mx = fmaxf(mx, t);
    }
    float s = 0.f;
    for (int g = 0; g < NG; ++g) { lg[g] = __expf(lg[g] - mx); s += lg[g]; }
    float inv = 1.0f / s;
    for (int g = 0; g < NG; ++g) gamma_out[(size_t)b * NG + g] = lg[g] * inv;
  }
}

// ========================= deterministic per-category batch permutation
__global__ __launch_bounds__(256) void perm_kernel(const float* __restrict__ dis,
                                                   float* __restrict__ disp) {
  int idx = blockIdx.x * 256 + threadIdx.x;  // < B*K*H
  int hcol = idx & 127;
  int k = (idx >> 7) & 7;
  int b = idx >> 10;
  const int am[KK] = {37, 93, 101, 171, 29, 55, 203, 89};  // odd -> bijective mod 256
  const int ad[KK] = {11, 200, 73, 148, 5, 97, 222, 39};
  int sb = (am[k] * b + ad[k]) & (BB - 1);
  disp[idx] = dis[((size_t)sb * KK + k) * HH + hcol];
}

// ================================================================= launcher
extern "C" void kernel_launch(void* const* d_in, const int* in_sizes, int n_in,
                              void* d_out, int out_size, void* d_ws, size_t ws_size,
                              hipStream_t stream) {
  const float* x        = (const float*)d_in[0];
  const float* enc_Wih  = (const float*)d_in[1];
  const float* enc_Whh  = (const float*)d_in[2];
  const float* enc_bih  = (const float*)d_in[3];
  const float* enc_bhh  = (const float*)d_in[4];
  const float* dec_Wih  = (const float*)d_in[5];
  const float* dec_Whh  = (const float*)d_in[6];
  const float* dec_bih  = (const float*)d_in[7];
  const float* dec_bhh  = (const float*)d_in[8];
  const float* out_W    = (const float*)d_in[9];
  const float* out_b    = (const float*)d_in[10];
  const float* protos   = (const float*)d_in[11];
  const float* lnz_w    = (const float*)d_in[12];
  const float* lnz_b    = (const float*)d_in[13];
  const float* lnp_w    = (const float*)d_in[14];
  const float* lnp_b    = (const float*)d_in[15];
  const float* lna_w    = (const float*)d_in[16];
  const float* lna_b    = (const float*)d_in[17];
  const float* beta     = (const float*)d_in[18];
  const float* e1W      = (const float*)d_in[19];
  const float* e1b      = (const float*)d_in[20];
  const float* e2W      = (const float*)d_in[21];
  const float* e2b      = (const float*)d_in[22];

  float* out = (float*)d_out;
  const size_t OFF_HENC = 0;
  const size_t OFF_OUT  = (size_t)BB * HH;                       // 32768
  const size_t OFF_Z    = OFF_OUT + (size_t)BB * SS * FF;        // +16777216
  const size_t OFF_GAM  = OFF_Z + (size_t)BB * LDIM;
  const size_t OFF_DIS  = OFF_GAM + (size_t)BB * NG;
  const size_t OFF_DISP = OFF_DIS + (size_t)BB * KK * HH;

  float* rc_ws   = (float*)d_ws;        // [256]
  float* prot_ws = rc_ws + BB;          // [8][128]

  const size_t enc_smem = (size_t)(384 * 128 * 2 + 4 * 16 * 128) * sizeof(__bf16);  // 212992 B
  const size_t dec_smem =
      (size_t)(384 * 128 * 2 + 128 * 128 + 3 * 16 * 128) * sizeof(__bf16);          // 241664 B

  enc_kernel<<<BB / 16, 256, enc_smem, stream>>>(x, enc_Wih, enc_Whh, enc_bih,
                                                 enc_bhh, out + OFF_HENC);
  dec_kernel<<<BB / 16, 256, dec_smem, stream>>>(out + OFF_HENC, dec_Wih,
                                                 dec_Whh, dec_bih, dec_bhh,
                                                 out_W, out_b, out + OFF_OUT);
  cos_kernel<<<BB, 256, 0, stream>>>(x, out + OFF_OUT, rc_ws);
  protln_kernel<<<KK, 128, 0, stream>>>(protos, lnp_w, lnp_b, prot_ws);
  cluster_kernel<<<BB, 128, 0, stream>>>(out + OFF_HENC, prot_ws, lnz_w, lnz_b,
                                         lna_w, lna_b, beta, e1W, e1b, e2W,
                                         e2b, rc_ws, out + OFF_Z, out + OFF_GAM,
                                         out + OFF_DIS);
  perm_kernel<<<(BB * KK * HH) / 256, 256, 0, stream>>>(out + OFF_DIS,
                                                        out + OFF_DISP);
  (void)in_sizes; (void)n_in; (void)out_size; (void)ws_size;
}
